// ReCo_ContrastiveLoss_18081812316876
// MI455X (gfx1250) — matrix operand order repaired and checked by reading, not compiled, not run on hardware
//
#include <hip/hip_runtime.h>
#include <cstdint>

typedef __attribute__((ext_vector_type(16))) _Float16 v16h;
typedef __attribute__((ext_vector_type(8)))  float    v8f;

#define SSP   524288      // z*x*y per batch
#define NPIX  1048576     // b*z*x*y
#define NCLS  4
#define NFEAT 64
#define NBLK  2048
#define BPX   512         // pixels per block (contiguous, stable order)
#define NQ    256
#define NNEG  256

// ---------------- threefry2x32-16 (16 rounds, 4 key injections) ----------------
__device__ __forceinline__ void tf2x32(unsigned k0, unsigned k1,
                                       unsigned x0, unsigned x1,
                                       unsigned& o0, unsigned& o1) {
  unsigned ks2 = k0 ^ k1 ^ 0x1BD11BDAu;
  x0 += k0; x1 += k1;
#define TFR(r) { x0 += x1; x1 = (x1 << (r)) | (x1 >> (32 - (r))); x1 ^= x0; }
  TFR(13) TFR(15) TFR(26) TFR(6)
  x0 += k1;  x1 += ks2 + 1u;
  TFR(17) TFR(29) TFR(16) TFR(24)
  x0 += ks2; x1 += k0 + 2u;
  TFR(13) TFR(15) TFR(26) TFR(6)
  x0 += k0;  x1 += k1 + 3u;
  TFR(17) TFR(29) TFR(16) TFR(24)
  x0 += k1;  x1 += ks2 + 4u;
#undef TFR
  o0 = x0; o1 = x1;
}

__device__ __forceinline__ float bits2f(unsigned b) {
  return __uint_as_float((b >> 9) | 0x3f800000u) - 1.0f;   // [0,1) with 23 random bits
}

// uniform(key, total)[idx] following jax threefry_2x32 split-in-half counter layout
__device__ __forceinline__ float unif_half(uint2 k, int idx, int total) {
  int half = total >> 1;
  unsigned o0, o1;
  if (idx < half) { tf2x32(k.x, k.y, (unsigned)idx, (unsigned)(idx + half), o0, o1); return bits2f(o0); }
  tf2x32(k.x, k.y, (unsigned)(idx - half), (unsigned)idx, o0, o1); return bits2f(o1);
}

__device__ __forceinline__ void split4(uint2 key, uint2 sub[4]) {
  unsigned a0,b0,a1,b1,a2,b2,a3,b3;
  tf2x32(key.x, key.y, 0u, 4u, a0, b0);
  tf2x32(key.x, key.y, 1u, 5u, a1, b1);
  tf2x32(key.x, key.y, 2u, 6u, a2, b2);
  tf2x32(key.x, key.y, 3u, 7u, a3, b3);
  sub[0] = make_uint2(a0, a1); sub[1] = make_uint2(a2, a3);
  sub[2] = make_uint2(b0, b1); sub[3] = make_uint2(b2, b3);
}

// rng = key(42); per class i: rng,kq,kc,kn = split(rng,4)
__device__ __forceinline__ void classKeys(int i, uint2& kq, uint2& kc, uint2& kn) {
  uint2 rng = make_uint2(0u, 42u);
  uint2 sub[4];
  for (int t = 0; t <= i; ++t) { split4(rng, sub); rng = sub[0]; }
  kq = sub[1]; kc = sub[2]; kn = sub[3];
}

// ---------------- hard flag: softmax(pred)[lab] < 1.0 ----------------
__device__ __forceinline__ int hard_flag(const float* __restrict__ pred, int n, int lab) {
  int bidx = n >> 19, sp = n & (SSP - 1);
  const float* pp = pred + (size_t)bidx * (NCLS * SSP) + sp;
  float x0 = pp[0], x1 = pp[SSP], x2 = pp[2 * SSP], x3 = pp[3 * SSP];
  float mx = fmaxf(fmaxf(x0, x1), fmaxf(x2, x3));
  float e0 = expf(x0 - mx), e1 = expf(x1 - mx), e2 = expf(x2 - mx), e3 = expf(x3 - mx);
  float sum = e0 + e1 + e2 + e3;
  float el = (lab == 0) ? e0 : (lab == 1) ? e1 : (lab == 2) ? e2 : e3;
  return (el / sum) < 1.0f;
}

// ============== K1: histograms + WMMA prototype GEMM (onehot^T @ feat) ==============
__global__ __launch_bounds__(256) void k_hist_proto(const float* __restrict__ pred,
                                                    const int* __restrict__ label,
                                                    const float* __restrict__ feat,
                                                    float* __restrict__ protoSum,
                                                    unsigned* __restrict__ bHist,
                                                    unsigned* __restrict__ bHistH) {
  __shared__ int      labs[BPX];
  __shared__ unsigned hist[NCLS], histH[NCLS];
  __shared__ float    sProto[256];
  int tid = threadIdx.x, blk = blockIdx.x;
  if (tid < NCLS) { hist[tid] = 0u; histH[tid] = 0u; }
  sProto[tid] = 0.0f;
  __syncthreads();
  int base = blk * BPX;

  for (int it = 0; it < 2; ++it) {
    int n = base + it * 256 + tid;
    int lab = label[n];
    labs[it * 256 + tid] = lab;
    atomicAdd(&hist[lab], 1u);
    if (hard_flag(pred, n, lab)) atomicAdd(&histH[lab], 1u);
  }
  __syncthreads();

  // WMMA: A = onehot (16 class rows, 4 live; 32 pixels K), B = feat (32 px x 16 feat tile)
  int wave = tid >> 5, lane = tid & 31;
  int M = lane & 15, h = lane >> 4;
  v8f acc[4] = {};
  for (int gi = 0; gi < 2; ++gi) {
    int g = wave + gi * 8;
    int pixBase = base + g * 32;
    int bidx = pixBase >> 19, sp = pixBase & (SSP - 1);
    v16h a;
#pragma unroll
    for (int j = 0; j < 16; ++j) {
      int v = j >> 1;
      int K = (h << 4) + j + (v & 0); // A: K = 16h + j (contiguous, matches B)
      a[j] = (labs[g * 32 + K] == M) ? (_Float16)1.0f : (_Float16)0.0f;
    }
    const float* fb = feat + (size_t)bidx * (NFEAT * SSP) + sp + (h << 4);
#pragma unroll
    for (int t = 0; t < 4; ++t) {
      const float* fp = fb + (size_t)((t << 4) + (lane & 15)) * SSP;
      __builtin_prefetch(fp + 256, 0, 3);   // next group, same channel
      v16h b;
#pragma unroll
      for (int j = 0; j < 16; ++j) b[j] = (_Float16)fp[j];   // B: lane=feature col, elem j=K 16h+j
      acc[t] = __builtin_amdgcn_wmma_f32_16x16x32_f16(false, a, false, b,
                                                      (short)0, acc[t], false, false);
    }
  }
  // D rows 0..3 live in VGPRs 0..3 of lanes 0..15 (col = lane)
  if (lane < 16) {
#pragma unroll
    for (int v = 0; v < 4; ++v) {
      atomicAdd(&sProto[v * 64 +  0 + lane], acc[0][v]);
      atomicAdd(&sProto[v * 64 + 16 + lane], acc[1][v]);
      atomicAdd(&sProto[v * 64 + 32 + lane], acc[2][v]);
      atomicAdd(&sProto[v * 64 + 48 + lane], acc[3][v]);
    }
  }
  __syncthreads();
  atomicAdd(&protoSum[tid], sProto[tid]);
  if (tid < NCLS) { bHist[blk * NCLS + tid] = hist[tid]; bHistH[blk * NCLS + tid] = histH[tid]; }
}

// ============== K2: block-histogram scans + prototypes + categorical logp ==============
__global__ __launch_bounds__(256) void k_scan(const float* __restrict__ protoSum,
                                              const unsigned* __restrict__ bHist,
                                              const unsigned* __restrict__ bHistH,
                                              unsigned* __restrict__ bOff,
                                              unsigned* __restrict__ bOffH,
                                              unsigned* __restrict__ counts,
                                              unsigned* __restrict__ hcounts,
                                              float* __restrict__ protoN,
                                              float* __restrict__ logp) {
  __shared__ unsigned buf[256];
  __shared__ unsigned sCnt[NCLS];
  __shared__ float    sRaw[256], sUnit[256], sNorm[NCLS], sSim[12];
  int tid = threadIdx.x;

  for (int a = 0; a < 8; ++a) {
    int c = a & 3; bool hs = a >= 4;
    const unsigned* src = hs ? bHistH : bHist;
    unsigned* dst = hs ? bOffH : bOff;
    unsigned vals[8], s = 0;
#pragma unroll
    for (int j = 0; j < 8; ++j) { vals[j] = src[(tid * 8 + j) * NCLS + c]; s += vals[j]; }
    buf[tid] = s; __syncthreads();
    for (int off = 1; off < 256; off <<= 1) {
      unsigned u = (tid >= off) ? buf[tid - off] : 0u;
      __syncthreads();
      buf[tid] += u;
      __syncthreads();
    }
    unsigned run = buf[tid] - s;
#pragma unroll
    for (int j = 0; j < 8; ++j) { dst[(tid * 8 + j) * NCLS + c] = run; run += vals[j]; }
    if (tid == 0) {
      unsigned total = buf[255];
      if (hs) hcounts[c] = total; else { counts[c] = total; sCnt[c] = total; }
    }
    __syncthreads();
  }

  { int c = tid >> 6; sRaw[tid] = protoSum[tid] / fmaxf((float)sCnt[c], 1.0f); }
  __syncthreads();
  if (tid < NCLS) {
    float ss = 0.0f;
    for (int j = 0; j < NFEAT; ++j) { float v = sRaw[tid * 64 + j]; ss += v * v; }
    sNorm[tid] = fmaxf(sqrtf(ss), 1e-8f);
  }
  __syncthreads();
  { int c = tid >> 6; sUnit[tid] = sRaw[tid] / sNorm[c]; protoN[tid] = sUnit[tid]; }
  __syncthreads();
  if (tid < 12) {
    int i = tid / 3, k = tid % 3, o = (i + 1 + k) & 3;
    float d = 0.0f;
    for (int j = 0; j < NFEAT; ++j) d += sUnit[i * 64 + j] * sUnit[o * 64 + j];
    sSim[tid] = d * 2.0f;   // sim / TAU
  }
  __syncthreads();
  if (tid < NCLS) {
    float s0 = sSim[tid * 3], s1 = sSim[tid * 3 + 1], s2 = sSim[tid * 3 + 2];
    float m = fmaxf(s0, fmaxf(s1, s2));
    float l = logf(expf(s0 - m) + expf(s1 - m) + expf(s2 - m)) + m;
    logp[tid * 3 + 0] = s0 - l; logp[tid * 3 + 1] = s1 - l; logp[tid * 3 + 2] = s2 - l;
  }
}

// ============== K3: stable counting-sort scatter (valid_order / hard_order) ==============
__global__ __launch_bounds__(256) void k_scatter(const float* __restrict__ pred,
                                                 const int* __restrict__ label,
                                                 const unsigned* __restrict__ bOff,
                                                 const unsigned* __restrict__ bOffH,
                                                 unsigned* __restrict__ clsList,
                                                 unsigned* __restrict__ hardList) {
  __shared__ unsigned bufA[256], bufB[256], bufC[256], bufD[256];
  __shared__ unsigned sRunA[NCLS], sRunH[NCLS];
  int tid = threadIdx.x, blk = blockIdx.x, base = blk * BPX;
  if (tid < NCLS) { sRunA[tid] = bOff[blk * NCLS + tid]; sRunH[tid] = bOffH[blk * NCLS + tid]; }
  __syncthreads();

  for (int it = 0; it < 2; ++it) {
    int n = base + it * 256 + tid;
    int lab = label[n];
    int hard = hard_flag(pred, n, lab);
    unsigned pA = ((lab == 0) ? 1u : 0u) | (((lab == 1) ? 1u : 0u) << 16);
    unsigned pB = ((lab == 2) ? 1u : 0u) | (((lab == 3) ? 1u : 0u) << 16);
    unsigned pC = hard ? pA : 0u, pD = hard ? pB : 0u;
    bufA[tid] = pA; bufB[tid] = pB; bufC[tid] = pC; bufD[tid] = pD;
    __syncthreads();
    for (int off = 1; off < 256; off <<= 1) {
      unsigned uA = 0, uB = 0, uC = 0, uD = 0;
      if (tid >= off) { uA = bufA[tid - off]; uB = bufB[tid - off]; uC = bufC[tid - off]; uD = bufD[tid - off]; }
      __syncthreads();
      bufA[tid] += uA; bufB[tid] += uB; bufC[tid] += uC; bufD[tid] += uD;
      __syncthreads();
    }
    unsigned eA = bufA[tid] - pA, eB = bufB[tid] - pB, eC = bufC[tid] - pC, eD = bufD[tid] - pD;
    unsigned rank, rankH;
    if      (lab == 0) { rank = eA & 0xffffu; rankH = eC & 0xffffu; }
    else if (lab == 1) { rank = eA >> 16;     rankH = eC >> 16;     }
    else if (lab == 2) { rank = eB & 0xffffu; rankH = eD & 0xffffu; }
    else               { rank = eB >> 16;     rankH = eD >> 16;     }
    clsList[(size_t)lab * NPIX + sRunA[lab] + rank] = (unsigned)n;
    if (hard) hardList[(size_t)lab * NPIX + sRunH[lab] + rankH] = (unsigned)n;
    __syncthreads();
    if (tid == 0) {
      unsigned tA = bufA[255], tB = bufB[255], tC = bufC[255], tD = bufD[255];
      sRunA[0] += tA & 0xffffu; sRunA[1] += tA >> 16;
      sRunA[2] += tB & 0xffffu; sRunA[3] += tB >> 16;
      sRunH[0] += tC & 0xffffu; sRunH[1] += tC >> 16;
      sRunH[2] += tD & 0xffffu; sRunH[3] += tD >> 16;
    }
    __syncthreads();
  }
}

// ============== K4: sample + normalize anchors, positive logit ==============
__global__ __launch_bounds__(256) void k_anchors(const float* __restrict__ feat,
                                                 const unsigned* __restrict__ hardList,
                                                 const unsigned* __restrict__ hcounts,
                                                 const float* __restrict__ protoN,
                                                 float* __restrict__ anchors,
                                                 float* __restrict__ posLog) {
  int tid = threadIdx.x, wave = tid >> 5, lane = tid & 31;
  int cls = blockIdx.x >> 5;
  int q = ((blockIdx.x & 31) << 3) + wave;
  uint2 kq, kc, kn; classKeys(cls, kq, kc, kn); (void)kc; (void)kn;

  float u = unif_half(kq, q, NQ);
  unsigned hc = hcounts[cls];
  unsigned pix = 0u;
  if (hc > 0u) {
    int t = (int)floorf(u * (float)hc);
    if (t > (int)hc - 1) t = (int)hc - 1;
    pix = hardList[(size_t)cls * NPIX + t];
  }
  int bidx = (int)(pix >> 19), sp = (int)(pix & (SSP - 1));
  const float* fp = feat + (size_t)bidx * (NFEAT * SSP) + sp;
  float f0 = fp[(size_t)(2 * lane) * SSP];
  float f1 = fp[(size_t)(2 * lane + 1) * SSP];
  float ss = f0 * f0 + f1 * f1;
  for (int m = 16; m >= 1; m >>= 1) ss += __shfl_xor(ss, m, 32);
  float scale = 1.0f / fmaxf(sqrtf(ss), 1e-8f);
  f0 *= scale; f1 *= scale;
  size_t ab = ((size_t)cls * NQ + q) * NFEAT + 2 * lane;
  anchors[ab] = f0; anchors[ab + 1] = f1;
  float p0 = protoN[cls * 64 + 2 * lane], p1 = protoN[cls * 64 + 2 * lane + 1];
  float d = f0 * p0 + f1 * p1;
  for (int m = 16; m >= 1; m >>= 1) d += __shfl_xor(d, m, 32);
  if (lane == 0) posLog[cls * NQ + q] = d * 2.0f;  // / TAU
}

// ============== K5: negatives, cosine logits, online log-sum-exp CE ==============
__global__ __launch_bounds__(256) void k_negloss(const float* __restrict__ feat,
                                                 const unsigned* __restrict__ clsList,
                                                 const unsigned* __restrict__ counts,
                                                 const float* __restrict__ logp,
                                                 const float* __restrict__ anchors,
                                                 const float* __restrict__ posLog,
                                                 float* __restrict__ lossSum) {
  int tid = threadIdx.x, wave = tid >> 5, lane = tid & 31;
  int cls = blockIdx.x >> 5;
  int q = ((blockIdx.x & 31) << 3) + wave;
  uint2 kq, kc, kn; classKeys(cls, kq, kc, kn); (void)kq;

  size_t ab = ((size_t)cls * NQ + q) * NFEAT + 2 * lane;
  float a0 = anchors[ab], a1 = anchors[ab + 1];
  float lp0 = logp[cls * 3 + 0], lp1 = logp[cls * 3 + 1], lp2 = logp[cls * 3 + 2];
  int o0 = (cls + 1) & 3, o1 = (cls + 2) & 3, o2 = (cls + 3) & 3;
  unsigned c0 = counts[o0], c1 = counts[o1], c2 = counts[o2];
  float pos = posLog[cls * NQ + q];
  float mx = pos, sm = 1.0f;

  for (int n = 0; n < NNEG; ++n) {
    int t3 = (q * NNEG + n) * 3;
    float u0 = fmaxf(unif_half(kc, t3,     196608), 1.17549435e-38f);
    float u1 = fmaxf(unif_half(kc, t3 + 1, 196608), 1.17549435e-38f);
    float u2 = fmaxf(unif_half(kc, t3 + 2, 196608), 1.17549435e-38f);
    float v0 = lp0 - logf(-logf(u0));
    float v1 = lp1 - logf(-logf(u1));
    float v2 = lp2 - logf(-logf(u2));
    int k = 0; float vb = v0;
    if (v1 > vb) { vb = v1; k = 1; }
    if (v2 > vb) { vb = v2; k = 2; }
    int nc = (k == 0) ? o0 : (k == 1) ? o1 : o2;
    unsigned cc = (k == 0) ? c0 : (k == 1) ? c1 : c2;

    float up = unif_half(kn, q * NNEG + n, 65536);
    unsigned pix = 0u;
    if (cc > 0u) {
      int t = (int)floorf(up * (float)cc);
      if (t > (int)cc - 1) t = (int)cc - 1;
      pix = clsList[(size_t)nc * NPIX + t];
    }
    int bidx = (int)(pix >> 19), sp = (int)(pix & (SSP - 1));
    const float* fp = feat + (size_t)bidx * (NFEAT * SSP) + sp;
    float f0 = fp[(size_t)(2 * lane) * SSP];
    float f1 = fp[(size_t)(2 * lane + 1) * SSP];
    float d = a0 * f0 + a1 * f1;
    float s = f0 * f0 + f1 * f1;
    for (int m = 16; m >= 1; m >>= 1) { d += __shfl_xor(d, m, 32); s += __shfl_xor(s, m, 32); }
    float logit = d / fmaxf(sqrtf(s), 1e-8f) * 2.0f;  // anchor already unit; / TAU
    if (logit > mx) { sm = sm * expf(mx - logit) + 1.0f; mx = logit; }
    else            { sm += expf(logit - mx); }
  }
  float lossq = (mx + logf(sm)) - pos;
  if (lane == 0) atomicAdd(&lossSum[cls], lossq);
}

// ============== K6: final combine ==============
__global__ void k_final(const unsigned* __restrict__ counts,
                        const unsigned* __restrict__ hcounts,
                        const float* __restrict__ lossSum,
                        float* __restrict__ out) {
  if (threadIdx.x == 0 && blockIdx.x == 0) {
    float tot = 0.0f; int vs = 0;
    for (int i = 0; i < NCLS; ++i) {
      if (hcounts[i] > 0u) tot += lossSum[i] / (float)NQ;
      if (counts[i] > 0u) vs++;
    }
    out[0] = tot / fmaxf((float)vs, 1.0f);
  }
}

// ============================== host launcher ==============================
extern "C" void kernel_launch(void* const* d_in, const int* in_sizes, int n_in,
                              void* d_out, int out_size, void* d_ws, size_t ws_size,
                              hipStream_t stream) {
  (void)in_sizes; (void)n_in; (void)out_size;
  const float* pred  = (const float*)d_in[0];
  const int*   label = (const int*)d_in[1];
  const float* feat  = (const float*)d_in[2];
  float* out = (float*)d_out;

  char* W = (char*)d_ws;
  float*    protoSum = (float*)(W + 0);            // 1024 B
  unsigned* cnts     = (unsigned*)(W + 1024);      // 16
  unsigned* hcnts    = (unsigned*)(W + 1088);      // 16
  float*    lossSum  = (float*)(W + 1152);         // 16
  float*    logp     = (float*)(W + 1216);         // 48
  float*    protoN   = (float*)(W + 1280);         // 1024
  float*    posLog   = (float*)(W + 2304);         // 4096
  float*    anchors  = (float*)(W + 8192);         // 262144
  unsigned* bHist    = (unsigned*)(W + 270336);    // 32768
  unsigned* bHistH   = (unsigned*)(W + 303104);    // 32768
  unsigned* bOff     = (unsigned*)(W + 335872);    // 32768
  unsigned* bOffH    = (unsigned*)(W + 368640);    // 32768
  unsigned* clsList  = (unsigned*)(W + 401408);                 // 16 MiB
  unsigned* hardList = (unsigned*)(W + 401408 + 16777216ull);   // 16 MiB
  const size_t NEED = 401408ull + 2ull * 16777216ull;
  if (ws_size < NEED) return;

  hipMemsetAsync(d_ws, 0, 2304, stream);   // zero protoSum/counts/loss accumulators

  k_hist_proto<<<NBLK, 256, 0, stream>>>(pred, label, feat, protoSum, bHist, bHistH);
  k_scan<<<1, 256, 0, stream>>>(protoSum, bHist, bHistH, bOff, bOffH, cnts, hcnts, protoN, logp);
  k_scatter<<<NBLK, 256, 0, stream>>>(pred, label, bOff, bOffH, clsList, hardList);
  k_anchors<<<128, 256, 0, stream>>>(feat, hardList, hcnts, protoN, anchors, posLog);
  k_negloss<<<128, 256, 0, stream>>>(feat, clsList, cnts, logp, anchors, posLog, lossSum);
  k_final<<<1, 1, 0, stream>>>(cnts, hcnts, lossSum, out);
}
